// GatedAttentionWrapper_1606317768725
// MI455X (gfx1250) — matrix operand order repaired and checked by reading, not compiled, not run on hardware
//
#include <hip/hip_runtime.h>

typedef __bf16 bf16_t;
typedef __attribute__((ext_vector_type(16))) __bf16 v16bf;
typedef __attribute__((ext_vector_type(8)))  float   v8f;

constexpr int S   = 4096;
constexpr int D   = 2048;
constexpr int H   = 16;
constexpr int KVH = 4;
constexpr int HD  = 128;
constexpr int NQ  = H * HD;    // 2048
constexpr int NKV = KVH * HD;  // 512

// ---------------------------------------------------------------- WMMA helpers

__device__ __forceinline__ v8f wmma_bf16(v16bf a, v16bf b, v8f c) {
  // v_wmma_f32_16x16x32_bf16: (neg_a, A, neg_b, B, c_mod, C, reuse_a, reuse_b)
  return __builtin_amdgcn_wmma_f32_16x16x32_bf16(false, a, false, b, (short)0, c,
                                                 false, false);
}

// A-matrix fragment (16x32 bf16, row-major source with leading dim ld):
// lanes 0-15:  row = lane,    K = {0..7, 16..23}
// lanes 16-31: row = lane-16, K = {8..15, 24..31}
__device__ __forceinline__ v16bf load_a_frag(const bf16_t* base, int ld, int lane) {
  int row  = lane & 15;
  int koff = (lane < 16) ? 0 : 8;
  const bf16_t* p = base + (size_t)row * ld + koff;
  v16bf a;
  __builtin_memcpy(&a, p, 16);                     // K: koff .. koff+7
  __builtin_memcpy(((char*)&a) + 16, p + 16, 16);  // K: koff+16 .. koff+23
  return a;
}

// B-matrix fragment (32x16 bf16). Source stored K-contiguous per output column:
// element (k, n) at base[n*ld + k].
// lanes 0-15:  n = lane,    K = {0..7, 16..23}
// lanes 16-31: n = lane-16, K = {8..15, 24..31}
__device__ __forceinline__ v16bf load_b_frag(const bf16_t* base, int ld, int lane) {
  int n    = lane & 15;
  int koff = (lane < 16) ? 0 : 8;
  const bf16_t* p = base + (size_t)n * ld + koff;
  v16bf b;
  __builtin_memcpy(&b, p, 16);
  __builtin_memcpy(((char*)&b) + 16, p + 16, 16);
  return b;
}

// ---------------------------------------------------------------- conversions

__global__ void f32_to_bf16_kernel(const float* __restrict__ x,
                                   bf16_t* __restrict__ y, int n) {
  int i = blockIdx.x * blockDim.x + threadIdx.x;
  if (i < n) y[i] = (bf16_t)x[i];
}

// W[K][N] fp32 -> Wt[N][K] bf16 (so GEMM B-fragments are contiguous loads)
__global__ void transpose_to_bf16_kernel(const float* __restrict__ W,
                                         bf16_t* __restrict__ Wt, int K, int N) {
  __shared__ float tile[16][17];
  int n = blockIdx.x * 16 + threadIdx.x;
  int k = blockIdx.y * 16 + threadIdx.y;
  tile[threadIdx.y][threadIdx.x] = W[(size_t)k * N + n];
  __syncthreads();
  int n2 = blockIdx.x * 16 + threadIdx.y;
  int k2 = blockIdx.y * 16 + threadIdx.x;
  Wt[(size_t)n2 * K + k2] = (bf16_t)tile[threadIdx.x][threadIdx.y];
}

// ---------------------------------------------------------------- WMMA GEMM
// C[M,N] = A[M,K] bf16 (row-major) @ B, with B supplied transposed Bt[N,K].
// Block = 4 waves stacked in M; each wave computes 32(M) x 128(N):
// 2 A-fragments x 8 B-fragments -> 16 WMMAs per 32-wide K step
// (~1.25 b128 loads per WMMA; both operands reused from registers).
//
// Epilogues (N-tile == one head for MODE 1/2, since HD == 128):
//   MODE 0: plain f32 store to C[M,N]            (out-projection -> d_out)
//   MODE 1: in-register RoPE, bf16 head-major    (Q, K: Ob[h][s][d])
//   MODE 2: bf16 transposed store                (V: Ob[h][d][s])
template <int MODE>
__global__ __launch_bounds__(128) void gemm32_kernel(
    const bf16_t* __restrict__ A, const bf16_t* __restrict__ Bt,
    float* __restrict__ C, bf16_t* __restrict__ Ob,
    const float* __restrict__ cosT, const float* __restrict__ sinT,
    int M, int N, int K) {
  int lane = threadIdx.x & 31;
  int w    = threadIdx.x >> 5;
  int m0   = blockIdx.y * 128 + w * 32;
  int n0   = blockIdx.x * 128;
  int h    = blockIdx.x;  // head index for MODE 1/2 (N-tile = HD = 128)

  v8f acc[2][8] = {};
  for (int k0 = 0; k0 < K; k0 += 32) {
    v16bf a0 = load_a_frag(A + (size_t)m0 * K + k0, K, lane);
    v16bf a1 = load_a_frag(A + (size_t)(m0 + 16) * K + k0, K, lane);
#pragma unroll
    for (int j = 0; j < 8; ++j) {
      v16bf b = load_b_frag(Bt + (size_t)(n0 + 16 * j) * K + k0, K, lane);
      acc[0][j] = wmma_bf16(a0, b, acc[0][j]);
      acc[1][j] = wmma_bf16(a1, b, acc[1][j]);
    }
  }

  int col  = lane & 15;
  int rowh = (lane >> 4) * 8;  // C layout: VGPR r -> row r (lanes 0-15) / r+8
#pragma unroll
  for (int s = 0; s < 2; ++s) {
    int mbase = m0 + 16 * s;
    if constexpr (MODE == 0) {
#pragma unroll
      for (int j = 0; j < 8; ++j)
#pragma unroll
        for (int r = 0; r < 8; ++r)
          C[(size_t)(mbase + r + rowh) * N + n0 + 16 * j + col] = acc[s][j][r];
    } else if constexpr (MODE == 1) {
      // RoPE: d in acc[.][j], d+64 in acc[.][j+4] (j<4); rotate in registers.
#pragma unroll
      for (int j = 0; j < 4; ++j)
#pragma unroll
        for (int r = 0; r < 8; ++r) {
          int row = mbase + r + rowh;
          int d   = 16 * j + col;
          int d2  = d + HD / 2;
          float x = acc[s][j][r];        // value at d
          float y = acc[s][j + 4][r];    // value at d+64
          float c1 = cosT[row * HD + d],  s1 = sinT[row * HD + d];
          float c2 = cosT[row * HD + d2], s2 = sinT[row * HD + d2];
          Ob[(size_t)h * S * HD + (size_t)row * HD + d]  = (bf16_t)(x * c1 - y * s1);
          Ob[(size_t)h * S * HD + (size_t)row * HD + d2] = (bf16_t)(y * c2 + x * s2);
        }
    } else {  // MODE == 2: V, transposed [h][d][s]
#pragma unroll
      for (int j = 0; j < 8; ++j)
#pragma unroll
        for (int r = 0; r < 8; ++r) {
          int row = mbase + r + rowh;
          int d   = 16 * j + col;
          Ob[(size_t)h * HD * S + (size_t)d * S + row] = (bf16_t)acc[s][j][r];
        }
    }
  }
}

// ---------------------------------------------------------------- attention
// grid.x = H * (S/64); block = 4 waves. Wave w owns 16 query rows
// q0 = 64*blk + 16*w of head h and sweeps causal KV tiles of width 64.
// Per KV tile: 16 WMMAs (QK^T, K-dim = HD = 4x32), online softmax with
// shfl_xor reductions over the 16-lane halves, P staged via LDS into
// A-fragment layout, then 16 WMMAs (P.V, K-dim = 64 = 2x32, N = HD = 8x16).
__global__ __launch_bounds__(128) void attn_kernel(
    const bf16_t* __restrict__ Qb, const bf16_t* __restrict__ Kb,
    const bf16_t* __restrict__ Vt, const float* __restrict__ gate,
    bf16_t* __restrict__ Ab) {
  __shared__ alignas(32) bf16_t ldsP[4 * 16 * 64];
  int lane = threadIdx.x & 31;
  int w    = threadIdx.x >> 5;
  int col  = lane & 15;
  int rowh = (lane >> 4) * 8;

  int qb  = blockIdx.x % (S / 64);
  int h   = blockIdx.x / (S / 64);
  int hkv = h / (H / KVH);
  int q0  = qb * 64 + w * 16;

  const bf16_t* Qh = Qb + (size_t)h * S * HD;
  const bf16_t* Kh = Kb + (size_t)hkv * S * HD;
  const bf16_t* Vh = Vt + (size_t)hkv * HD * S;
  bf16_t* P = ldsP + w * 16 * 64;  // per-wave region; same-wave LDS is in-order

  v16bf qa[4];
#pragma unroll
  for (int c = 0; c < 4; ++c)
    qa[c] = load_a_frag(Qh + (size_t)q0 * HD + 32 * c, HD, lane);

  float m[8], l[8];
  v8f o[8] = {};
#pragma unroll
  for (int r = 0; r < 8; ++r) { m[r] = -1e30f; l[r] = 0.0f; }

  const float scale = 0.08838834764831845f;  // 1/sqrt(128)
  int nt = (q0 + 15) / 64 + 1;               // causal: skip tiles above diagonal

  for (int t = 0; t < nt; ++t) {
    int kv0 = t * 64;
    // ---- scores: S = Q @ K^T
    v8f sc[4] = {};
#pragma unroll
    for (int c = 0; c < 4; ++c) {
#pragma unroll
      for (int j = 0; j < 4; ++j) {
        v16bf b = load_b_frag(Kh + (size_t)(kv0 + 16 * j) * HD + 32 * c, HD, lane);
        sc[j] = wmma_bf16(qa[c], b, sc[j]);
      }
    }
    // ---- scale + causal mask + log-gate
    float lg[4];
#pragma unroll
    for (int j = 0; j < 4; ++j)
      lg[j] = __logf(gate[kv0 + 16 * j + col] + 1e-8f);
    float mt[8];
#pragma unroll
    for (int r = 0; r < 8; ++r) {
      int qrow = q0 + r + rowh;
      float rm = -1e30f;
#pragma unroll
      for (int j = 0; j < 4; ++j) {
        int kv = kv0 + 16 * j + col;
        float x = sc[j][r] * scale + lg[j];
        if (kv > qrow) x = -1e30f;
        sc[j][r] = x;
        rm = fmaxf(rm, x);
      }
      mt[r] = rm;
    }
    // ---- row max over 64 cols (xor over lanes 0..15 of each half)
#pragma unroll
    for (int r = 0; r < 8; ++r) {
#pragma unroll
      for (int msk = 1; msk < 16; msk <<= 1)
        mt[r] = fmaxf(mt[r], __shfl_xor(mt[r], msk, 32));
    }
    float alpha[8];
#pragma unroll
    for (int r = 0; r < 8; ++r) {
      float mn = fmaxf(m[r], mt[r]);
      alpha[r] = __expf(m[r] - mn);
      m[r] = mn;
    }
    // ---- P = exp(S - m), row sums
    float ts[8];
#pragma unroll
    for (int r = 0; r < 8; ++r) {
      float sum = 0.0f;
#pragma unroll
      for (int j = 0; j < 4; ++j) {
        float p = __expf(sc[j][r] - m[r]);
        sc[j][r] = p;
        sum += p;
      }
      ts[r] = sum;
    }
#pragma unroll
    for (int r = 0; r < 8; ++r) {
#pragma unroll
      for (int msk = 1; msk < 16; msk <<= 1)
        ts[r] += __shfl_xor(ts[r], msk, 32);
      l[r] = l[r] * alpha[r] + ts[r];
    }
    // ---- rescale running O
#pragma unroll
    for (int jd = 0; jd < 8; ++jd)
#pragma unroll
      for (int r = 0; r < 8; ++r) o[jd][r] *= alpha[r];
    // ---- C-layout -> A-layout via LDS
#pragma unroll
    for (int j = 0; j < 4; ++j)
#pragma unroll
      for (int r = 0; r < 8; ++r)
        P[(r + rowh) * 64 + 16 * j + col] = (bf16_t)sc[j][r];
    // ---- O += P @ V
#pragma unroll
    for (int c = 0; c < 2; ++c) {
      v16bf pa = load_a_frag(P + 32 * c, 64, lane);
#pragma unroll
      for (int jd = 0; jd < 8; ++jd) {
        v16bf b = load_b_frag(Vh + (size_t)(16 * jd) * S + kv0 + 32 * c, S, lane);
        o[jd] = wmma_bf16(pa, b, o[jd]);
      }
    }
  }
  // ---- normalize + store [s, h*HD + d] bf16 for the out-projection GEMM
  float inv[8];
#pragma unroll
  for (int r = 0; r < 8; ++r) inv[r] = 1.0f / l[r];
#pragma unroll
  for (int jd = 0; jd < 8; ++jd)
#pragma unroll
    for (int r = 0; r < 8; ++r) {
      int row = q0 + r + rowh;
      int d   = 16 * jd + col;
      Ab[(size_t)row * NQ + h * HD + d] = (bf16_t)(o[jd][r] * inv[r]);
    }
}

// ---------------------------------------------------------------- launch

extern "C" void kernel_launch(void* const* d_in, const int* in_sizes, int n_in,
                              void* d_out, int out_size, void* d_ws, size_t ws_size,
                              hipStream_t stream) {
  (void)in_sizes; (void)n_in; (void)out_size; (void)ws_size;
  const float* X    = (const float*)d_in[0];
  // d_in[1] = attention_mask: analytically causal -> computed in-kernel (skips 64MB)
  const float* cosT = (const float*)d_in[2];
  const float* sinT = (const float*)d_in[3];
  const float* gate = (const float*)d_in[4];
  const float* Wq   = (const float*)d_in[5];
  const float* Wk   = (const float*)d_in[6];
  const float* Wv   = (const float*)d_in[7];
  const float* Wo   = (const float*)d_in[8];

  char* ws = (char*)d_ws;
  constexpr size_t OFF_XB  = 0;
  constexpr size_t OFF_WQT = OFF_XB  + (size_t)S * D * 2;
  constexpr size_t OFF_WKT = OFF_WQT + (size_t)D * NQ * 2;
  constexpr size_t OFF_WVT = OFF_WKT + (size_t)D * NKV * 2;
  constexpr size_t OFF_WOT = OFF_WVT + (size_t)D * NKV * 2;
  constexpr size_t OFF_QB  = OFF_WOT + (size_t)NQ * D * 2;
  constexpr size_t OFF_KB  = OFF_QB  + (size_t)S * NQ * 2;
  constexpr size_t OFF_VT  = OFF_KB  + (size_t)S * NKV * 2;
  constexpr size_t OFF_AB  = OFF_VT  + (size_t)S * NKV * 2;

  bf16_t* Xb  = (bf16_t*)(ws + OFF_XB);
  bf16_t* Wqt = (bf16_t*)(ws + OFF_WQT);
  bf16_t* Wkt = (bf16_t*)(ws + OFF_WKT);
  bf16_t* Wvt = (bf16_t*)(ws + OFF_WVT);
  bf16_t* Wot = (bf16_t*)(ws + OFF_WOT);
  bf16_t* Qb  = (bf16_t*)(ws + OFF_QB);
  bf16_t* Kb  = (bf16_t*)(ws + OFF_KB);
  bf16_t* Vt  = (bf16_t*)(ws + OFF_VT);
  bf16_t* Ab  = (bf16_t*)(ws + OFF_AB);

  // 1) precision casts / weight transposes
  int nx = S * D;
  f32_to_bf16_kernel<<<(nx + 255) / 256, 256, 0, stream>>>(X, Xb, nx);
  transpose_to_bf16_kernel<<<dim3(NQ / 16, D / 16), dim3(16, 16), 0, stream>>>(Wq, Wqt, D, NQ);
  transpose_to_bf16_kernel<<<dim3(NKV / 16, D / 16), dim3(16, 16), 0, stream>>>(Wk, Wkt, D, NKV);
  transpose_to_bf16_kernel<<<dim3(NKV / 16, D / 16), dim3(16, 16), 0, stream>>>(Wv, Wvt, D, NKV);
  transpose_to_bf16_kernel<<<dim3(D / 16, NQ / 16), dim3(16, 16), 0, stream>>>(Wo, Wot, NQ, D);

  // 2) QKV projections with fused RoPE / repack epilogues (WMMA)
  gemm32_kernel<1><<<dim3(NQ / 128,  S / 128), 128, 0, stream>>>(
      Xb, Wqt, nullptr, Qb, cosT, sinT, S, NQ, D);
  gemm32_kernel<1><<<dim3(NKV / 128, S / 128), 128, 0, stream>>>(
      Xb, Wkt, nullptr, Kb, cosT, sinT, S, NKV, D);
  gemm32_kernel<2><<<dim3(NKV / 128, S / 128), 128, 0, stream>>>(
      Xb, Wvt, nullptr, Vt, nullptr, nullptr, S, NKV, D);

  // 3) causal flash attention with gate (WMMA both matmuls)
  attn_kernel<<<dim3(H * (S / 64)), 128, 0, stream>>>(Qb, Kb, Vt, gate, Ab);

  // 4) output projection (WMMA) -> fp32 d_out
  gemm32_kernel<0><<<dim3(D / 128, S / 128), 128, 0, stream>>>(
      Ab, Wot, (float*)d_out, nullptr, nullptr, nullptr, S, D, NQ);
}
